// AttLMModel_51427938402747
// MI455X (gfx1250) — compile-verified, matrix operand order checked by reading, not run on hardware
//
#include <hip/hip_runtime.h>
#include <cstdint>
#include <cstddef>

// ---------------------------------------------------------------------------
// Problem constants
// ---------------------------------------------------------------------------
#define NVOC  50257
#define NIN   1024
#define NHID  1024
#define H2    2048
#define TT    1024
#define BB    64

// ---------------------------------------------------------------------------
// Vector types for WMMA
// ---------------------------------------------------------------------------
typedef __attribute__((ext_vector_type(16))) __bf16        v16bf;
typedef __attribute__((ext_vector_type(8)))  float         v8f;
typedef __attribute__((ext_vector_type(2)))  float         v2f;
typedef __attribute__((ext_vector_type(4)))  unsigned int  v4u;
typedef __attribute__((ext_vector_type(8)))  unsigned int  v8u;

__device__ __forceinline__ v8f v8f_zero() {
  v8f z = {0.f,0.f,0.f,0.f,0.f,0.f,0.f,0.f};
  return z;
}

// Manual f32 -> bf16 (RNE) and back, avoiding __bf16 scalar arithmetic.
__device__ __forceinline__ unsigned short f2bf(float x) {
  unsigned int u = __float_as_uint(x);
  u += 0x7FFFu + ((u >> 16) & 1u);
  return (unsigned short)(u >> 16);
}
__device__ __forceinline__ float bf2f(unsigned short h) {
  return __uint_as_float(((unsigned int)h) << 16);
}

// bf16 16x16x32 A/B fragment loader. LDS layout is [row][k] (row = M for A,
// row = N for B since we store B [n][k]-major). Per ISA 7.12.2:
// lane L: row = L%16 (applied by caller via row pointer), h = L/16,
// VGPR0..3 hold K = [8h, 8h+7], VGPR4..7 hold K = [16+8h, 16+8h+7].
__device__ __forceinline__ v16bf ld_frag16(const unsigned short* rowp, int k0, int h) {
  struct U { v4u a; v4u b; } u;
  u.a = *(const v4u*)(rowp + k0 + 8 * h);
  u.b = *(const v4u*)(rowp + k0 + 16 + 8 * h);
  return __builtin_bit_cast(v16bf, u);
}

// f32 16x16x4 A/B fragment: lane L: row=L%16, h=L/16, {K=2h, K=2h+1}.
__device__ __forceinline__ v2f ld_frag32(const float* rowp, int k0, int h) {
  return *(const v2f*)(rowp + k0 + 2 * h);
}

__device__ __forceinline__ float sigmoidf_(float x) {
  return 1.0f / (1.0f + expf(-x));
}

// Async global -> LDS (per-lane addresses) + wait.
__device__ __forceinline__ void async_b32(unsigned int lds_off, const void* gaddr) {
  unsigned long long g = (unsigned long long)(uintptr_t)gaddr;
  asm volatile("global_load_async_to_lds_b32 %0, %1, off"
               :: "v"(lds_off), "v"(g) : "memory");
}
__device__ __forceinline__ void wait_async0() {
  asm volatile("s_wait_asynccnt 0" ::: "memory");
}

// ---------------------------------------------------------------------------
// Tensor Data Mover: generic 2-D tile load (D# groups 0/1, f32 elements).
//   tile0 x tile1 elements, row stride stride0 (elements), optional LDS
//   padding (pad codes per ISA 8.4) to synthesize padded LDS pitches,
//   tensor_dim OOB zero-fill for ragged edges.
// ---------------------------------------------------------------------------
__device__ __forceinline__ void tdm_issue_2d(const void* gbase, unsigned int lds_off,
                                             unsigned int tile0, unsigned int tile1,
                                             unsigned int td0, unsigned int td1,
                                             unsigned long long stride0,
                                             unsigned int pad_en, unsigned int pad_int,
                                             unsigned int pad_amt) {
  unsigned long long ga = (unsigned long long)(uintptr_t)gbase;
  v4u g0;
  g0.x = 1u;                                           // count=1, user mode
  g0.y = lds_off;                                      // LDS dest byte address
  g0.z = (unsigned int)ga;                             // global addr [31:0]
  g0.w = ((unsigned int)(ga >> 32) & 0x1FFFFFFu) | (2u << 30);  // [56:32]|type=2
  v8u g1;
  g1[0] = 0x00020000u | (pad_en << 20) | (pad_int << 22) | (pad_amt << 25);
  g1[1] = (td0 & 0xFFFFu) << 16;
  g1[2] = (td0 >> 16) | ((td1 & 0xFFFFu) << 16);
  g1[3] = (td1 >> 16) | (tile0 << 16);
  g1[4] = tile1;                                       // tile_dim1 | tile_dim2=0
  g1[5] = (unsigned int)stride0;
  g1[6] = (unsigned int)(stride0 >> 32) & 0xFFFFu;
  g1[7] = 0u;
  asm volatile("tensor_load_to_lds %0, %1" :: "s"(g0), "s"(g1) : "memory");
}

// 4x4 register transpose: src is 16-bit k-major (row stride NHID elems),
// dst is [n][k]-major LDS with pitch `pitch`. b64 loads + b64 stores.
__device__ __forceinline__ void stage_tr4_bf16(const unsigned short* __restrict__ src,
                                               unsigned short* dst,
                                               int gk, int gn, int lk, int ln4,
                                               int pitch) {
  unsigned long long q[4];
#pragma unroll
  for (int d = 0; d < 4; ++d)
    q[d] = *(const unsigned long long*)(src + (size_t)(gk + d) * NHID + gn);
#pragma unroll
  for (int d = 0; d < 4; ++d) {
    unsigned long long o = ((q[0] >> (16 * d)) & 0xFFFFull)
                         | (((q[1] >> (16 * d)) & 0xFFFFull) << 16)
                         | (((q[2] >> (16 * d)) & 0xFFFFull) << 32)
                         | (((q[3] >> (16 * d)) & 0xFFFFull) << 48);
    *(unsigned long long*)(dst + (size_t)(ln4 + d) * pitch + lk) = o;
  }
}

// ---------------------------------------------------------------------------
// K0: split attUa (f32) into bf16 hi/lo planes (4 MB -> 2+2 MB, L2-resident)
// ---------------------------------------------------------------------------
__global__ __launch_bounds__(256) void conv_split_kernel(
    const float* __restrict__ src, unsigned short* __restrict__ hi,
    unsigned short* __restrict__ lo, int n) {
  int i = blockIdx.x * 256 + threadIdx.x;
  if (i < n) {
    float x = src[i];
    unsigned short h = f2bf(x);
    hi[i] = h;
    lo[i] = f2bf(x - bf2f(h));
  }
}

// ---------------------------------------------------------------------------
// K1: hWa[b][n] = sum_k h_prev[b][k] * attWa[k][n]   (tiny, L2-resident)
// ---------------------------------------------------------------------------
__global__ __launch_bounds__(256) void hwa_kernel(
    const float* __restrict__ h_prev, const float* __restrict__ attWa,
    float* __restrict__ hwa) {
  __shared__ float hrow[NHID];
  const int b = blockIdx.x, nc = blockIdx.y;
  for (int i = threadIdx.x; i < NHID; i += 256) hrow[i] = h_prev[b * NHID + i];
  __syncthreads();
  const int n = nc * 256 + threadIdx.x;
  float acc = 0.f;
  for (int k = 0; k < NHID; ++k) acc = fmaf(hrow[k], attWa[k * NHID + n], acc);
  hwa[b * NHID + n] = acc;
}

// ---------------------------------------------------------------------------
// K2: score[r] = attV . tanh(hWa[b] + mem[r] @ attUa)   r = t*B + b
// fp32-via-bf16x3 WMMA. A-panel (32 rows x 1024 K): one TDM load of the raw
// f32 panel (memory_pool read exactly once from HBM), converted in LDS to
// bf16 hi/lo. attUa hi/lo streamed from L2 in 128x64 chunks with a 4x4
// register transpose into [n][k]-major LDS.
// ---------------------------------------------------------------------------
#define SC_PA   1032   // A lds pitch (ushort elems), 2064 B = 129*16
#define SC_PB   136    // B lds pitch (ushort elems), 272 B = 17*16
#define SC_LDS  (2*32*SC_PA*2 + 2*64*SC_PB*2 + 32*4 + 32*1024*4)

__global__ __launch_bounds__(256) void score_kernel(
    const float* __restrict__ mem, const unsigned short* __restrict__ ua_hi,
    const unsigned short* __restrict__ ua_lo, const float* __restrict__ hwa,
    const float* __restrict__ attV, float* __restrict__ score) {
  extern __shared__ char lds_raw[];
  unsigned short* aHi = (unsigned short*)lds_raw;      // [32][SC_PA]
  unsigned short* aLo = aHi + 32 * SC_PA;
  unsigned short* bHi = aLo + 32 * SC_PA;              // [64][SC_PB]
  unsigned short* bLo = bHi + 64 * SC_PB;
  float* scoreAcc     = (float*)(bLo + 64 * SC_PB);    // [32]
  float* araw         = scoreAcc + 32;                 // [32*1024] staging

  const int tid = threadIdx.x;
  const int wave = tid >> 5, lane = tid & 31;
  const int mt = wave >> 2, nt = wave & 3;             // 2 M-tiles x 4 N-tiles
  const int h = lane >> 4, ln = lane & 15;
  const int rowbase = blockIdx.x * 32;

  if (tid < 32) {
    scoreAcc[tid] = 0.f;
    // --- TDM: stage raw A panel (32 rows x 1024 f32, contiguous) ---
    tdm_issue_2d(mem + (size_t)rowbase * NHID, (unsigned int)(uintptr_t)araw,
                 1024, 32, 1u << 30, 1u << 30, 1024, 0, 0, 0);
    __builtin_amdgcn_s_wait_tensorcnt(0);
  }
  __syncthreads();

  // --- convert A panel to bf16 hi/lo (float4 in, packed u64 out) ---
  for (int i = 0; i < 32; ++i) {
    int e4 = tid + i * 256;                            // 8192 float4
    int r = e4 >> 8, k4 = (e4 & 255) * 4;
    float4 x = ((const float4*)araw)[e4];
    float xs[4] = {x.x, x.y, x.z, x.w};
    unsigned long long hq = 0, lq = 0;
#pragma unroll
    for (int j = 0; j < 4; ++j) {
      unsigned short hh = f2bf(xs[j]);
      unsigned short ll = f2bf(xs[j] - bf2f(hh));
      hq |= (unsigned long long)hh << (16 * j);
      lq |= (unsigned long long)ll << (16 * j);
    }
    *(unsigned long long*)(aHi + r * SC_PA + k4) = hq;
    *(unsigned long long*)(aLo + r * SC_PA + k4) = lq;
  }
  __syncthreads();

  float spart[8];
#pragma unroll
  for (int r = 0; r < 8; ++r) spart[r] = 0.f;

#pragma unroll 1
  for (int nb = 0; nb < 16; ++nb) {                    // N blocks of 64
    v8f acc = v8f_zero();
#pragma unroll 1
    for (int kc = 0; kc < 8; ++kc) {                   // K chunks of 128
      __syncthreads();
      // stage attUa chunk transposed: 4x4 register transpose, hi & lo planes
      for (int i = 0; i < 2; ++i) {
        int id = tid + i * 256;                        // 512 4x4 tiles
        int tn = id & 15, tk = id >> 4;
        int gk = kc * 128 + tk * 4, gn = nb * 64 + tn * 4;
        stage_tr4_bf16(ua_hi, bHi, gk, gn, tk * 4, tn * 4, SC_PB);
        stage_tr4_bf16(ua_lo, bLo, gk, gn, tk * 4, tn * 4, SC_PB);
      }
      __syncthreads();
      const unsigned short* arH = aHi + (mt * 16 + ln) * SC_PA + kc * 128;
      const unsigned short* arL = aLo + (mt * 16 + ln) * SC_PA + kc * 128;
      const unsigned short* brH = bHi + (nt * 16 + ln) * SC_PB;
      const unsigned short* brL = bLo + (nt * 16 + ln) * SC_PB;
#pragma unroll
      for (int ks = 0; ks < 4; ++ks) {
        const int k0 = ks * 32;
        v16bf ah = ld_frag16(arH, k0, h);
        v16bf al = ld_frag16(arL, k0, h);
        v16bf bh = ld_frag16(brH, k0, h);
        v16bf bl = ld_frag16(brL, k0, h);
        acc = __builtin_amdgcn_wmma_f32_16x16x32_bf16(false, ah, false, bh,
                                                      (short)0, acc, false, false);
        acc = __builtin_amdgcn_wmma_f32_16x16x32_bf16(false, ah, false, bl,
                                                      (short)0, acc, false, false);
        acc = __builtin_amdgcn_wmma_f32_16x16x32_bf16(false, al, false, bh,
                                                      (short)0, acc, false, false);
      }
    }
    // post-process: + hWa, tanh, * attV, reduce over 16 cols
    const int col = nb * 64 + nt * 16 + ln;
    const float av = attV[col];
#pragma unroll
    for (int r = 0; r < 8; ++r) {
      int row = rowbase + mt * 16 + r + 8 * h;
      float v = tanhf(acc[r] + hwa[(row & (BB - 1)) * NHID + col]) * av;
      v += __shfl_xor(v, 1, 32);
      v += __shfl_xor(v, 2, 32);
      v += __shfl_xor(v, 4, 32);
      v += __shfl_xor(v, 8, 32);
      spart[r] += v;
    }
  }
  __syncthreads();
  if (ln == 0) {
#pragma unroll
    for (int r = 0; r < 8; ++r)
      atomicAdd(&scoreAcc[mt * 16 + r + 8 * h], spart[r]);
  }
  __syncthreads();
  if (tid < 32) score[rowbase + tid] = scoreAcc[tid];
}

// ---------------------------------------------------------------------------
// K3: softmax over T per batch column
// ---------------------------------------------------------------------------
__global__ __launch_bounds__(256) void softmax_kernel(
    const float* __restrict__ score, float* __restrict__ attw) {
  __shared__ float red[256];
  const int b = blockIdx.x;
  float v[4];
  float mx = -1e30f;
#pragma unroll
  for (int i = 0; i < 4; ++i) {
    v[i] = score[(threadIdx.x + i * 256) * BB + b];
    mx = fmaxf(mx, v[i]);
  }
  red[threadIdx.x] = mx;
  __syncthreads();
  for (int s = 128; s > 0; s >>= 1) {
    if (threadIdx.x < s) red[threadIdx.x] = fmaxf(red[threadIdx.x], red[threadIdx.x + s]);
    __syncthreads();
  }
  mx = red[0];
  __syncthreads();
  float sum = 0.f;
#pragma unroll
  for (int i = 0; i < 4; ++i) { v[i] = expf(v[i] - mx); sum += v[i]; }
  red[threadIdx.x] = sum;
  __syncthreads();
  for (int s = 128; s > 0; s >>= 1) {
    if (threadIdx.x < s) red[threadIdx.x] += red[threadIdx.x + s];
    __syncthreads();
  }
  const float inv = 1.f / red[0];
#pragma unroll
  for (int i = 0; i < 4; ++i)
    attw[(threadIdx.x + i * 256) * BB + b] = v[i] * inv;
}

// ---------------------------------------------------------------------------
// K4: contextVec[b][n] = sum_t attw[t,b] * mem[t,b,n]
// TDM streams memory_pool (2-D tiles 256 f32 x 64 t-rows, stride B*NHID),
// double-buffered in LDS; the strided attw column is gathered with per-lane
// async-to-LDS b32.
// ---------------------------------------------------------------------------
#define CTX_LDS (TT * 4 + 2 * 64 * 256 * 4)

__global__ __launch_bounds__(256) void context_kernel(
    const float* __restrict__ mem, const float* __restrict__ attw,
    float* __restrict__ ctx) {
  extern __shared__ char lds_raw[];
  float* wcol = (float*)lds_raw;                       // [1024]
  float* buf0 = wcol + TT;                             // [64*256]
  float* buf1 = buf0 + 64 * 256;                       // [64*256]
  const int tid = threadIdx.x;
  const int b = blockIdx.x, nc = blockIdx.y;

  // gather softmax weight column via async-to-LDS (per-lane strided)
  {
    const unsigned int woff = (unsigned int)(uintptr_t)wcol;
    for (int i = 0; i < 4; ++i) {
      int t = tid + i * 256;
      async_b32(woff + (unsigned int)t * 4, attw + (size_t)t * BB + b);
    }
  }

  const unsigned int off0 = (unsigned int)(uintptr_t)buf0;
  const unsigned int off1 = (unsigned int)(uintptr_t)buf1;

  if (tid < 32)
    tdm_issue_2d(mem + (size_t)b * NHID + nc * 256, off0,
                 256, 64, 1u << 30, 1u << 30, (unsigned long long)BB * NHID,
                 0, 0, 0);
  wait_async0();

  float acc = 0.f;
#pragma unroll 1
  for (int c = 0; c < 16; ++c) {
    if (tid < 32) {
      if (c + 1 < 16) {
        tdm_issue_2d(mem + ((size_t)(c + 1) * 64 * BB + b) * NHID + nc * 256,
                     ((c + 1) & 1) ? off1 : off0,
                     256, 64, 1u << 30, 1u << 30, (unsigned long long)BB * NHID,
                     0, 0, 0);
        __builtin_amdgcn_s_wait_tensorcnt(1);          // chunk c complete
      } else {
        __builtin_amdgcn_s_wait_tensorcnt(0);
      }
    }
    __syncthreads();
    const float* bufc = (c & 1) ? buf1 : buf0;
    const float* wc = wcol + c * 64;
#pragma unroll 8
    for (int r = 0; r < 64; ++r) acc = fmaf(wc[r], bufc[r * 256 + tid], acc);
    __syncthreads();
  }
  ctx[b * NHID + nc * 256 + tid] = acc;
}

// ---------------------------------------------------------------------------
// K5: X = [emb | h_prev | context]  (64 x 3072), emb = enc_w gather
// ---------------------------------------------------------------------------
__global__ __launch_bounds__(256) void buildx_kernel(
    const int* __restrict__ inp, const float* __restrict__ enc_w,
    const float* __restrict__ h_prev, const float* __restrict__ ctx,
    float* __restrict__ X) {
  int id = blockIdx.x * 256 + threadIdx.x;             // 64*3072
  int b = id / 3072, k = id % 3072;
  float v;
  if (k < NIN)        v = enc_w[(size_t)inp[b] * NIN + k];
  else if (k < 2048)  v = h_prev[b * NHID + (k - NIN)];
  else                v = ctx[b * NHID + (k - 2048)];
  X[id] = v;
}

// ---------------------------------------------------------------------------
// K6: gates = X @ [W_ih;W_hh] + b_lstm   (64 x 3072 x 8192), f32 WMMA.
// X chunk: one TDM per K-chunk, with LDS padding synthesizing the 132-dword
// pitch (interval 128 dw, amount 4 dw); overlapped with the manual 4x4
// register-transpose staging of the W chunk.
// ---------------------------------------------------------------------------
#define GT_PX   132
#define GT_PW   132
#define GT_LDS  (64 * GT_PX * 4 + 128 * GT_PW * 4)

__global__ __launch_bounds__(256) void gates_kernel(
    const float* __restrict__ X, const float* __restrict__ W_ih,
    const float* __restrict__ W_hh, const float* __restrict__ b_lstm,
    float* __restrict__ gates) {
  extern __shared__ char lds_raw[];
  float* Xl = (float*)lds_raw;                         // [64][GT_PX]
  float* Wl = Xl + 64 * GT_PX;                         // [128][GT_PW]
  const int tid = threadIdx.x;
  const int wave = tid >> 5, lane = tid & 31;
  const int h = lane >> 4, ln = lane & 15;
  const int nb = blockIdx.x;                           // 64 col-blocks of 128

  v8f acc[4];
#pragma unroll
  for (int m = 0; m < 4; ++m) acc[m] = v8f_zero();

  const unsigned int xloff = (unsigned int)(uintptr_t)Xl;

#pragma unroll 1
  for (int kc = 0; kc < 24; ++kc) {                    // K chunks of 128
    __syncthreads();
    if (tid < 32)                                      // TDM: X chunk 64x128
      tdm_issue_2d(X + kc * 128, xloff, 128, 64, 1u << 30, 1u << 30, 3072,
                   1, 6, 3);                           // pad -> pitch 132
    // W chunk transposed to [n][k] via 4x4 register transpose (overlaps TDM)
    for (int i = 0; i < 4; ++i) {
      int id = tid + i * 256;                          // 1024 4x4 tiles
      int tn = id & 31, tk = id >> 5;
      int gk = kc * 128 + tk * 4, gn = nb * 128 + tn * 4;
      const float* wbase = (gk < NIN) ? (W_ih + (size_t)gk * 8192)
                                      : (W_hh + (size_t)(gk - NIN) * 8192);
      float q[4][4];
#pragma unroll
      for (int d = 0; d < 4; ++d) {
        float4 t = *(const float4*)(wbase + (size_t)d * 8192 + gn);
        q[d][0] = t.x; q[d][1] = t.y; q[d][2] = t.z; q[d][3] = t.w;
      }
#pragma unroll
      for (int d = 0; d < 4; ++d) {
        float4 o = {q[0][d], q[1][d], q[2][d], q[3][d]};
        *(float4*)(Wl + (size_t)(tn * 4 + d) * GT_PW + tk * 4) = o;
      }
    }
    if (tid < 32) __builtin_amdgcn_s_wait_tensorcnt(0);
    __syncthreads();
    const float* bp = Wl + (wave * 16 + ln) * GT_PW;
    const float* ap = Xl + ln * GT_PX;
#pragma unroll 8
    for (int ks = 0; ks < 32; ++ks) {
      const int k0 = ks * 4;
      v2f bf = ld_frag32(bp, k0, h);
#pragma unroll
      for (int m = 0; m < 4; ++m) {
        v2f af = ld_frag32(ap + m * 16 * GT_PX, k0, h);
        acc[m] = __builtin_amdgcn_wmma_f32_16x16x4_f32(false, af, false, bf,
                                                       (short)0, acc[m], false, false);
      }
    }
  }
  const int col = nb * 128 + wave * 16 + ln;
  const float bias = b_lstm[col];
#pragma unroll
  for (int m = 0; m < 4; ++m)
#pragma unroll
    for (int r = 0; r < 8; ++r) {
      int row = m * 16 + r + 8 * h;
      gates[(size_t)row * 8192 + col] = acc[m][r] + bias;
    }
}

// ---------------------------------------------------------------------------
// K7: LSTM elementwise: ct, ht (ht also copied to ws for decode)
// ---------------------------------------------------------------------------
__global__ __launch_bounds__(256) void lstm_kernel(
    const float* __restrict__ gates, const float* __restrict__ c_prev,
    float* __restrict__ ct_out, float* __restrict__ ht_out,
    float* __restrict__ ht_ws) {
  int id = blockIdx.x * 256 + threadIdx.x;             // 64*2048
  int b = id >> 11, j = id & 2047;
  const float* g = gates + (size_t)b * 8192;
  float ig = sigmoidf_(g[j]);
  float fg = sigmoidf_(g[2048 + j]);
  float gg = tanhf(g[4096 + j]);
  float og = sigmoidf_(g[6144 + j]);
  float ct = fg * c_prev[id] + ig * gg;
  float ht = og * tanhf(ct);
  ct_out[id] = ct;
  if (j < NHID) {
    ht_out[b * NHID + j] = ht;
    ht_ws[b * NHID + j] = ht;
  }
}

// ---------------------------------------------------------------------------
// K8: decoded = ht @ dec_w.T + dec_b   (64 x 1024 x 50257), f32 WMMA.
// Full ht panel: one TDM with pad -> 1032-dword pitch. dec_w B chunks: one
// TDM per K-chunk (pad -> 68-dword pitch) with hardware OOB zero-fill for
// the vocabulary tail (tensor_dim1 = NVOC - vb).
// ---------------------------------------------------------------------------
#define DC_PA   1032
#define DC_PB   68
#define DC_LDS  (64 * DC_PA * 4 + 128 * DC_PB * 4)

__global__ __launch_bounds__(256) void decode_kernel(
    const float* __restrict__ ht, const float* __restrict__ dec_w,
    const float* __restrict__ dec_b, float* __restrict__ out) {
  extern __shared__ char lds_raw[];
  float* Al = (float*)lds_raw;                         // [64][DC_PA]
  float* Bl = Al + 64 * DC_PA;                         // [128][DC_PB]
  const int tid = threadIdx.x;
  const int wave = tid >> 5, lane = tid & 31;
  const int h = lane >> 4, ln = lane & 15;
  const int vb = blockIdx.x * 128;

  const unsigned int aloff = (unsigned int)(uintptr_t)Al;
  const unsigned int bloff = (unsigned int)(uintptr_t)Bl;

  if (tid < 32)                                        // TDM: full ht panel
    tdm_issue_2d(ht, aloff, 1024, 64, 1u << 30, 1u << 30, 1024,
                 1, 7, 1);                             // pad -> pitch 1032

  v8f acc[4];
#pragma unroll
  for (int m = 0; m < 4; ++m) acc[m] = v8f_zero();

#pragma unroll 1
  for (int kc = 0; kc < 16; ++kc) {                    // K chunks of 64
    __syncthreads();
    if (tid < 32) {                                    // TDM: B chunk [v][k]
      tdm_issue_2d(dec_w + (size_t)vb * NHID + kc * 64, bloff,
                   64, 128, 1024 - kc * 64, (unsigned int)(NVOC - vb), 1024,
                   1, 5, 3);                           // pad -> pitch 68
      __builtin_amdgcn_s_wait_tensorcnt(0);
    }
    __syncthreads();
    const float* bp = Bl + (wave * 16 + ln) * DC_PB;
    // padded-pitch A addressing: +2*(kc>>2) from the TDM pad every 256 dw
    const float* ap = Al + ln * DC_PA + kc * 64 + 2 * (kc >> 2);
#pragma unroll 4
    for (int ks = 0; ks < 16; ++ks) {
      const int k0 = ks * 4;
      v2f bf = ld_frag32(bp, k0, h);
#pragma unroll
      for (int m = 0; m < 4; ++m) {
        v2f af = ld_frag32(ap + m * 16 * DC_PA, k0, h);
        acc[m] = __builtin_amdgcn_wmma_f32_16x16x4_f32(false, af, false, bf,
                                                       (short)0, acc[m], false, false);
      }
    }
  }
  const int v = vb + wave * 16 + ln;
  if (v < NVOC) {
    const float bias = dec_b[v];
#pragma unroll
    for (int m = 0; m < 4; ++m)
#pragma unroll
      for (int r = 0; r < 8; ++r) {
        int row = m * 16 + r + 8 * h;
        out[(size_t)row * NVOC + v] = acc[m][r] + bias;
      }
  }
}

// ---------------------------------------------------------------------------
// Host launcher
// ---------------------------------------------------------------------------
extern "C" void kernel_launch(void* const* d_in, const int* in_sizes, int n_in,
                              void* d_out, int out_size, void* d_ws, size_t ws_size,
                              hipStream_t stream) {
  const int*   inputs = (const int*)d_in[0];
  const float* mem    = (const float*)d_in[1];
  const float* h_prev = (const float*)d_in[2];
  const float* c_prev = (const float*)d_in[3];
  const float* enc_w  = (const float*)d_in[4];
  const float* attWa  = (const float*)d_in[5];
  const float* attUa  = (const float*)d_in[6];
  const float* attV   = (const float*)d_in[7];
  const float* W_ih   = (const float*)d_in[8];
  const float* W_hh   = (const float*)d_in[9];
  const float* b_lstm = (const float*)d_in[10];
  const float* dec_w  = (const float*)d_in[11];
  const float* dec_b  = (const float*)d_in[12];
  (void)in_sizes; (void)n_in; (void)out_size; (void)ws_size;

  float* out_dec = (float*)d_out;                      // (64, 50257)
  float* out_ht  = out_dec + (size_t)BB * NVOC;        // (1, 64, 1024)
  float* out_ct  = out_ht + BB * NHID;                 // (1, 64, 2048)

  char* ws = (char*)d_ws;
  float* hwa   = (float*)ws;  ws += (size_t)BB * NHID * 4;
  float* score = (float*)ws;  ws += (size_t)TT * BB * 4;
  float* attw  = (float*)ws;  ws += (size_t)TT * BB * 4;
  float* ctx   = (float*)ws;  ws += (size_t)BB * NHID * 4;
  float* X     = (float*)ws;  ws += (size_t)BB * 3072 * 4;
  float* gatesb= (float*)ws;  ws += (size_t)BB * 8192 * 4;
  float* ht_ws = (float*)ws;  ws += (size_t)BB * NHID * 4;
  unsigned short* ua_hi = (unsigned short*)ws; ws += (size_t)NHID * NHID * 2;
  unsigned short* ua_lo = (unsigned short*)ws; ws += (size_t)NHID * NHID * 2;

  // >64KB dynamic LDS kernels
  hipFuncSetAttribute((const void*)score_kernel,
                      hipFuncAttributeMaxDynamicSharedMemorySize, SC_LDS);
  hipFuncSetAttribute((const void*)context_kernel,
                      hipFuncAttributeMaxDynamicSharedMemorySize, CTX_LDS);
  hipFuncSetAttribute((const void*)gates_kernel,
                      hipFuncAttributeMaxDynamicSharedMemorySize, GT_LDS);
  hipFuncSetAttribute((const void*)decode_kernel,
                      hipFuncAttributeMaxDynamicSharedMemorySize, DC_LDS);

  conv_split_kernel<<<(NHID * NHID) / 256, 256, 0, stream>>>(attUa, ua_hi, ua_lo,
                                                             NHID * NHID);
  hwa_kernel<<<dim3(BB, 4), 256, 0, stream>>>(h_prev, attWa, hwa);
  score_kernel<<<(TT * BB) / 32, 256, SC_LDS, stream>>>(mem, ua_hi, ua_lo, hwa,
                                                        attV, score);
  softmax_kernel<<<BB, 256, 0, stream>>>(score, attw);
  context_kernel<<<dim3(BB, 4), 256, CTX_LDS, stream>>>(mem, attw, ctx);
  buildx_kernel<<<(BB * 3072) / 256, 256, 0, stream>>>(inputs, enc_w, h_prev,
                                                       ctx, X);
  gates_kernel<<<64, 256, GT_LDS, stream>>>(X, W_ih, W_hh, b_lstm, gatesb);
  lstm_kernel<<<(BB * H2) / 256, 256, 0, stream>>>(gatesb, c_prev, out_ct,
                                                   out_ht, ht_ws);
  decode_kernel<<<(NVOC + 127) / 128, 256, DC_LDS, stream>>>(ht_ws, dec_w,
                                                             dec_b, out_dec);
}
